// VoxelFieldL2Loss_67001489817678
// MI455X (gfx1250) — matrix operand order repaired and checked by reading, not compiled, not run on hardware
//
#include <hip/hip_runtime.h>
#include <hip/hip_bf16.h>

typedef __attribute__((ext_vector_type(16))) _Float16 v16h;
typedef __attribute__((ext_vector_type(8)))  float    v8f;

constexpr int   kBatch = 4;
constexpr int   kNP    = 2048;
constexpr int   kNV    = 16384;
constexpr float kSigma = 0.04f;

__launch_bounds__(256)
__global__ void VoxelFieldL2Loss_kernel(const float* __restrict__ y_hat,
                                        const float* __restrict__ y,
                                        const float* __restrict__ vc,
                                        float* __restrict__ out)
{
    // Stage the entire y[b] slab (2048 points x 6 floats = 48KB) in LDS.
    __shared__ float s_y[kNP * 6];

    const int tilesPerBatch = kNV / 128;            // 8 waves x 16 voxels per block
    const int b      = blockIdx.x / tilesPerBatch;
    const int vblock = blockIdx.x % tilesPerBatch;

    {
        const float4* src = reinterpret_cast<const float4*>(y + (size_t)b * kNP * 6);
        float4*       dst = reinterpret_cast<float4*>(s_y);
        const int n4 = kNP * 6 / 4;
        for (int i = threadIdx.x; i < n4; i += blockDim.x) dst[i] = src[i];
    }
    __syncthreads();

    const int lane  = threadIdx.x & 31;   // wave32
    const int wave  = threadIdx.x >> 5;
    const int h     = lane >> 4;          // half-wave select
    const int m     = lane & 15;          // A row (voxel) == B column (channel)
    const int vbase = vblock * 128 + wave * 16;

    // Each lane owns one voxel (one A-matrix row) -> its center stays in VGPRs.
    const float vx = vc[(vbase + m) * 3 + 0];
    const float vy = vc[(vbase + m) * 3 + 1];
    const float vz = vc[(vbase + m) * 3 + 2];

    const bool  haveB = (m < 3);          // only B columns 0..2 carry pw
    const int   bcol  = haveB ? m : 0;
    const float bmask = haveB ? 1.0f : 0.0f;

    // exp(-d2/(2*sigma^2)) == exp2(d2 * expScale)
    const float expScale = -1.4426950408889634f / (2.0f * kSigma * kSigma);

    v8f acc = {};

    for (int pbase = 0; pbase < kNP; pbase += 32) {
        v16h amat, bmat;
#pragma unroll
        for (int j = 0; j < 16; ++j) {
            // A-matrix (f16 16x32) K index for element j in this lane:
            //   h=0: K = {0..7, 16..23}; h=1: K = {8..15, 24..31}
            const int ka = j + 8 * h + ((j >> 3) << 3);
            const int pa = pbase + ka;
            const float px = s_y[pa * 6 + 0];
            const float py = s_y[pa * 6 + 1];
            const float pz = s_y[pa * 6 + 2];
            const float dx = vx - px, dy = vy - py, dz = vz - pz;
            const float d2 = dx * dx + dy * dy + dz * dz;
            amat[j] = (_Float16)__builtin_amdgcn_exp2f(d2 * expScale);

            // B-matrix (f16 32x16) K index, same as A: h=0: K = {0..7, 16..23}; h=1: K = {8..15, 24..31}
            const int kb = j + 8 * h + ((j >> 3) << 3);
            const int pb = pbase + kb;
            bmat[j] = (_Float16)(s_y[pb * 6 + 3 + bcol] * bmask);
        }
        // D = A(gaussian) x B(pw) + C ; accumulates field_gt tile (16 voxels x 3 channels)
        acc = __builtin_amdgcn_wmma_f32_16x16x32_f16(false, amat, false, bmat,
                                                     (short)0, acc, false, false);
    }

    // C/D layout: acc[r] (lane L) = field[vbase + r + 8h, n = L%16]
    if (haveB) {
        float partial = 0.0f;
#pragma unroll
        for (int r = 0; r < 8; ++r) {
            const int   vv  = vbase + r + 8 * h;
            const float ref = y_hat[((size_t)b * 3 + m) * kNV + vv];
            const float d   = ref - acc[r];
            partial += d * d;
        }
        __hip_atomic_fetch_add(out, partial * (1.0f / kBatch),
                               __ATOMIC_RELAXED, __HIP_MEMORY_SCOPE_AGENT);
    }
}

extern "C" void kernel_launch(void* const* d_in, const int* in_sizes, int n_in,
                              void* d_out, int out_size, void* d_ws, size_t ws_size,
                              hipStream_t stream) {
    (void)in_sizes; (void)n_in; (void)out_size; (void)d_ws; (void)ws_size;
    const float* y_hat = (const float*)d_in[0];
    const float* y     = (const float*)d_in[1];
    const float* vc    = (const float*)d_in[2];
    float* out = (float*)d_out;

    hipMemsetAsync(d_out, 0, sizeof(float), stream);   // capture-safe zero of the scalar

    dim3 grid(kBatch * (kNV / 128));                   // 512 blocks, 8 waves each
    VoxelFieldL2Loss_kernel<<<grid, dim3(256), 0, stream>>>(y_hat, y, vc, out);
}
